// PerTokenSelector_4827543240912
// MI455X (gfx1250) — compile-verified, matrix operand order checked by reading, not run on hardware
//
#include <hip/hip_runtime.h>
#include <hip/hip_bf16.h>
#include <math.h>

typedef __attribute__((ext_vector_type(16))) __bf16 v16bf;
typedef __attribute__((ext_vector_type(8)))  __bf16 bf16x8;
typedef __attribute__((ext_vector_type(8)))  float  v8f;
typedef __attribute__((ext_vector_type(4)))  float  f32x4;

#define D 2048
#define E 64
#define RTEMP (1.0f / 8.0f)
#define LN_EPS 1e-5f
#define UEPS 1e-8f
#define TOK_PER_BLOCK 128
#define LSTRIDE 65   // 64 experts + 1 pad -> conflict-free LDS scans

// ---------------- Kernel 1: unit-normalize prototypes -> bf16 in workspace ----
__global__ void __launch_bounds__(256) proto_norm_kernel(
    const float* __restrict__ proto, __bf16* __restrict__ pn) {
  __shared__ float s_part[8];
  __shared__ float s_rnorm;
  const int e   = blockIdx.x;
  const int tid = threadIdx.x;
  const float* row = proto + (size_t)e * D;

  float q = 0.f;
#pragma unroll
  for (int i = 0; i < D / 256; ++i) {
    float v = row[tid + i * 256];
    q += v * v;
  }
#pragma unroll
  for (int off = 16; off; off >>= 1) q += __shfl_xor(q, off, 32);
  const int lane = tid & 31, wave = tid >> 5;
  if (lane == 0) s_part[wave] = q;
  __syncthreads();
  if (tid == 0) {
    float s = 0.f;
#pragma unroll
    for (int w = 0; w < 8; ++w) s += s_part[w];
    s_rnorm = 1.0f / fmaxf(sqrtf(s), UEPS);
  }
  __syncthreads();
  const float rn = s_rnorm;
#pragma unroll
  for (int i = 0; i < D / 256; ++i) {
    const int idx = tid + i * 256;
    pn[(size_t)e * D + idx] = (__bf16)(row[idx] * rn);
  }
}

// ---------------- Kernel 2: fused LN + bf16 WMMA GEMM + top-2 router ---------
__global__ void __launch_bounds__(256) router_kernel(
    const float* __restrict__ x, const __bf16* __restrict__ pn,
    float* __restrict__ out_w, int* __restrict__ out_i) {
  __shared__ float s_mu[TOK_PER_BLOCK];
  __shared__ float s_rstd[TOK_PER_BLOCK];
  __shared__ float s_logits[8 * 16 * LSTRIDE];

  const int tid     = threadIdx.x;
  const int lane    = tid & 31;
  const int wave    = tid >> 5;
  const int blockTok = blockIdx.x * TOK_PER_BLOCK;
  const int waveTok  = blockTok + wave * 16;

  // ---- Phase 1: LayerNorm stats for this wave's 16 tokens (coalesced b128) --
  for (int t = 0; t < 16; ++t) {
    const f32x4* xr = (const f32x4*)(x + (size_t)(waveTok + t) * D);
    float s = 0.f, q = 0.f;
#pragma unroll
    for (int i = 0; i < D / 128; ++i) {
      f32x4 v = xr[i * 32 + lane];
      s += v.x + v.y + v.z + v.w;
      q += v.x * v.x + v.y * v.y + v.z * v.z + v.w * v.w;
    }
#pragma unroll
    for (int off = 16; off; off >>= 1) {
      s += __shfl_xor(s, off, 32);
      q += __shfl_xor(q, off, 32);
    }
    if (lane == 0) {
      float mu  = s * (1.0f / D);
      float var = q * (1.0f / D) - mu * mu;
      s_mu[wave * 16 + t]   = mu;
      s_rstd[wave * 16 + t] = rsqrtf(var + LN_EPS);
    }
  }
  __syncthreads();

  // ---- Phase 2: C[16,64] = LN(x)[16,2048] x pn^T via v_wmma_f32_16x16x32_bf16
  const int   rowInBlk = wave * 16 + (lane & 15);
  const float mu   = s_mu[rowInBlk];
  const float rstd = s_rstd[rowInBlk];
  const float* xrow = x + (size_t)(blockTok + rowInBlk) * D;
  const int kb = (lane < 16) ? 0 : 8;  // per-lane K base (ISA 16-bit A layout)

  const __bf16* brow0 = pn + (size_t)( 0 + (lane & 15)) * D;
  const __bf16* brow1 = pn + (size_t)(16 + (lane & 15)) * D;
  const __bf16* brow2 = pn + (size_t)(32 + (lane & 15)) * D;
  const __bf16* brow3 = pn + (size_t)(48 + (lane & 15)) * D;

  v8f acc0 = {}, acc1 = {}, acc2 = {}, acc3 = {};

  for (int k0 = 0; k0 < D; k0 += 32) {
    // A fragment: elems 0..7 -> K=kb..kb+7 ; elems 8..15 -> K=kb+16..kb+23
    union { f32x4 v[4]; float f[16]; } ax;
    const f32x4* xp = (const f32x4*)(xrow + k0 + kb);
    ax.v[0] = xp[0]; ax.v[1] = xp[1];   // K = kb+0..7
    ax.v[2] = xp[4]; ax.v[3] = xp[5];   // K = kb+16..23
    v16bf a;
#pragma unroll
    for (int t = 0; t < 16; ++t) a[t] = (__bf16)((ax.f[t] - mu) * rstd);

    union { v16bf v; bf16x8 h[2]; } b0, b1, b2, b3;
    { const bf16x8* p = (const bf16x8*)(brow0 + k0 + kb); b0.h[0] = p[0]; b0.h[1] = p[2]; }
    { const bf16x8* p = (const bf16x8*)(brow1 + k0 + kb); b1.h[0] = p[0]; b1.h[1] = p[2]; }
    { const bf16x8* p = (const bf16x8*)(brow2 + k0 + kb); b2.h[0] = p[0]; b2.h[1] = p[2]; }
    { const bf16x8* p = (const bf16x8*)(brow3 + k0 + kb); b3.h[0] = p[0]; b3.h[1] = p[2]; }

    acc0 = __builtin_amdgcn_wmma_f32_16x16x32_bf16(false, a, false, b0.v, (short)0, acc0, false, false);
    acc1 = __builtin_amdgcn_wmma_f32_16x16x32_bf16(false, a, false, b1.v, (short)0, acc1, false, false);
    acc2 = __builtin_amdgcn_wmma_f32_16x16x32_bf16(false, a, false, b2.v, (short)0, acc2, false, false);
    acc3 = __builtin_amdgcn_wmma_f32_16x16x32_bf16(false, a, false, b3.v, (short)0, acc3, false, false);
  }

  // ---- Phase 3: spill tile to LDS (C layout: m = r + 8*(lane/16), n = lane%16)
  float* lt = s_logits + wave * 16 * LSTRIDE;
  const int mh = (lane >> 4) * 8;
  const int n  = lane & 15;
#pragma unroll
  for (int r = 0; r < 8; ++r) {
    lt[(mh + r) * LSTRIDE +  0 + n] = acc0[r];
    lt[(mh + r) * LSTRIDE + 16 + n] = acc1[r];
    lt[(mh + r) * LSTRIDE + 32 + n] = acc2[r];
    lt[(mh + r) * LSTRIDE + 48 + n] = acc3[r];
  }
  __syncthreads();

  // ---- Phase 4: top-2 of logits; softmax denom cancels under renormalization
  if (lane < 16) {
    const float* lr = lt + lane * LSTRIDE;
    float l1 = lr[0]; int i1 = 0;
#pragma unroll
    for (int k = 1; k < E; ++k) { float v = lr[k]; if (v > l1) { l1 = v; i1 = k; } }
    float l2 = -INFINITY; int i2 = 0;
#pragma unroll
    for (int k = 0; k < E; ++k) {
      if (k == i1) continue;
      float v = lr[k]; if (v > l2) { l2 = v; i2 = k; }
    }
    const float en = __expf((l2 - l1) * RTEMP);   // <= 1
    const float w1 = 1.0f / (1.0f + en);
    const float w2 = en * w1;
    const int token = waveTok + lane;
    out_w[token * 2 + 0] = w1;
    out_w[token * 2 + 1] = w2;
    out_i[token * 2 + 0] = i1;
    out_i[token * 2 + 1] = i2;
  }
}

extern "C" void kernel_launch(void* const* d_in, const int* in_sizes, int n_in,
                              void* d_out, int out_size, void* d_ws, size_t ws_size,
                              hipStream_t stream) {
  const float* x     = (const float*)d_in[0];
  const float* proto = (const float*)d_in[1];
  __bf16* pn = (__bf16*)d_ws;                 // 64*2048*2 = 256 KB of scratch

  const int ntok = in_sizes[0] / D;           // 4*4096 = 16384
  float* out_w = (float*)d_out;               // first output: [ntok,2] f32
  int*   out_i = (int*)((float*)d_out + (size_t)ntok * 2); // second: [ntok,2] i32

  proto_norm_kernel<<<E, 256, 0, stream>>>(proto, pn);
  router_kernel<<<ntok / TOK_PER_BLOCK, 256, 0, stream>>>(x, pn, out_w, out_i);
}
